// GaussianSplattingDecoder_82265803588034
// MI455X (gfx1250) — matrix operand order repaired and checked by reading, not compiled, not run on hardware
//
#include <hip/hip_runtime.h>
#include <hip/hip_bf16.h>
#include <math.h>

typedef __attribute__((ext_vector_type(16))) _Float16 v16h;
typedef __attribute__((ext_vector_type(8)))  float    v8f;

#define N_GAUSS   2048
#define N_CLASSES 17
#define N_VOX     80000
#define K_TILE    32
#define N_KTILES  (N_GAUSS / K_TILE)      // 64
#define PARAMS_F  12                       // floats per gaussian record
#define PARAMS_BYTES (N_GAUSS * PARAMS_F * 4)            // 98304

#define HALF_LOG2E 0.7213475204444817f    // 0.5 * log2(e)
#define LOG2E      1.4426950408889634f

// ---------------------------------------------------------------------------
// Prep 1: fold activations + quat->inverse-covariance into 12-float records,
// pre-scaled so the hot loop is pure FMA + exp2:
//   weight = exp2(-(lopac + c00'*d0^2 + c11'*d1^2 + c22'*d2^2
//                   + c01'*d0*d1 + c02'*d0*d2 + c12'*d1*d2))
// with diag scaled by 0.5*log2e, off-diag by log2e (absorbs the 2x), and
// lopac = -log2(opacity) seeding the FMA chain (absorbs the opacity mul).
//   [m0 m1 m2 | c00' c01' c02' c11' c12' c22' | lopac | pad pad]
// ---------------------------------------------------------------------------
__global__ void prep_params_kernel(const float* __restrict__ props,
                                   float* __restrict__ params) {
  int g = blockIdx.x * blockDim.x + threadIdx.x;
  if (g >= N_GAUSS) return;
  const float* p = props + (size_t)g * 28;

  float m0 = p[0], m1 = p[1], m2 = p[2];

  float s0 = fmaxf(log1pf(__expf(p[3])), 1e-4f);
  float s1 = fmaxf(log1pf(__expf(p[4])), 1e-4f);
  float s2 = fmaxf(log1pf(__expf(p[5])), 1e-4f);
  float is0 = 1.0f / fmaxf(s0 * s0, 1e-8f);
  float is1 = 1.0f / fmaxf(s1 * s1, 1e-8f);
  float is2 = 1.0f / fmaxf(s2 * s2, 1e-8f);

  float qw = p[6], qx = p[7], qy = p[8], qz = p[9];
  float qn = sqrtf(qw * qw + qx * qx + qy * qy + qz * qz);
  float qi = 1.0f / fmaxf(qn, 1e-8f);
  qw *= qi; qx *= qi; qy *= qi; qz *= qi;
  float r00 = 1.f - 2.f * (qy * qy + qz * qz);
  float r01 = 2.f * (qx * qy - qw * qz);
  float r02 = 2.f * (qx * qz + qw * qy);
  float r10 = 2.f * (qx * qy + qw * qz);
  float r11 = 1.f - 2.f * (qx * qx + qz * qz);
  float r12 = 2.f * (qy * qz - qw * qx);
  float r20 = 2.f * (qx * qz - qw * qy);
  float r21 = 2.f * (qy * qz + qw * qx);
  float r22 = 1.f - 2.f * (qx * qx + qy * qy);

  float c00 = r00 * is0 * r00 + r01 * is1 * r01 + r02 * is2 * r02;
  float c01 = r00 * is0 * r10 + r01 * is1 * r11 + r02 * is2 * r12;
  float c02 = r00 * is0 * r20 + r01 * is1 * r21 + r02 * is2 * r22;
  float c11 = r10 * is0 * r10 + r11 * is1 * r11 + r12 * is2 * r12;
  float c12 = r10 * is0 * r20 + r11 * is1 * r21 + r12 * is2 * r22;
  float c22 = r20 * is0 * r20 + r21 * is1 * r21 + r22 * is2 * r22;

  float opac  = 1.0f / (1.0f + __expf(-p[10]));
  float lopac = -__log2f(opac);

  float* o = params + (size_t)g * PARAMS_F;
  o[0] = m0;               o[1] = m1;               o[2] = m2;
  o[3] = c00 * HALF_LOG2E; o[4] = c01 * LOG2E;      o[5] = c02 * LOG2E;
  o[6] = c11 * HALF_LOG2E; o[7] = c12 * LOG2E;      o[8] = c22 * HALF_LOG2E;
  o[9] = lopac;            o[10] = 0.0f;            o[11] = 0.0f;
}

// ---------------------------------------------------------------------------
// Prep 2: pre-swizzle semantics into the exact WMMA B-matrix lane layout.
// B is 32x16 (KxN) f16: lanes 0-15 hold K=0..7,16..23 (N=lane), lanes 16-31
// hold K=8..15,24..31 (N=lane-16); element j<8 -> K 8h+j, j>=8 -> K 16+8h+j-8.
// Buffer: Bbuf[ktile][ctile][lane][16 halves].
// ---------------------------------------------------------------------------
__global__ void prep_bmat_kernel(const float* __restrict__ props,
                                 _Float16* __restrict__ Bbuf) {
  int tid = blockIdx.x * blockDim.x + threadIdx.x;   // 0 .. 4095
  if (tid >= N_KTILES * 2 * 32) return;
  int L   = tid & 31;
  int ct  = (tid >> 5) & 1;
  int t   = tid >> 6;
  int cls = ct * 16 + (L & 15);
  int kbase = t * K_TILE + (L >> 4) * 8;
  _Float16* dst = Bbuf + (size_t)tid * 16;
#pragma unroll
  for (int j = 0; j < 16; ++j) {
    float val = 0.0f;
    if (cls < N_CLASSES) val = props[(size_t)(kbase + j + ((j < 8) ? 0 : 8)) * 28 + 11 + cls];
    dst[j] = (_Float16)val;
  }
}

// ---------------------------------------------------------------------------
// Main: one wave = 16 voxels (M). Loop K over 64 tiles of 32 gaussians.
// VALU+trans generate the f16 A tile (branchless FMA chain + exp2), the
// matrix pipe does the class accumulation: 2 WMMAs/tile. LDS param staging
// is double-buffered -> one split workgroup barrier per K-tile.
// ---------------------------------------------------------------------------
__global__ __launch_bounds__(256)
void splat_kernel(const float* __restrict__ params,
                  const _Float16* __restrict__ Bbuf,
                  const float* __restrict__ vox,
                  float* __restrict__ out) {
  __shared__ float4 stage4[2][K_TILE * 3];   // 2 x 1.5 KB double buffer

  const int tid   = threadIdx.x;
  const int lane  = tid & 31;
  const int wave  = tid >> 5;
  const int tile  = blockIdx.x * 8 + wave;     // 625 * 8 = 5000 tiles
  const int vbase = tile * 16;

  const int m     = lane & 15;                 // voxel row this lane feeds
  const int khalf = lane >> 4;                 // which K half of the A tile
  const int kh8   = khalf * 8;

  const int vme = vbase + m;
  const float vx = vox[vme * 3 + 0];
  const float vy = vox[vme * 3 + 1];
  const float vz = vox[vme * 3 + 2];

  const float4* params4 = (const float4*)params;

  v8f acc0 = {};   // classes 0..15
  v8f acc1 = {};   // classes 16..31 (only 16 live)

  // prime buffer 0
  if (tid < K_TILE * 3) stage4[0][tid] = params4[tid];
  __syncthreads();

#pragma unroll 2
  for (int t = 0; t < N_KTILES; ++t) {
    const int buf = t & 1;

    // stage next K-chunk into the other buffer (flies under the VALU work)
    if (t + 1 < N_KTILES && tid < K_TILE * 3)
      stage4[buf ^ 1][tid] = params4[(t + 1) * (K_TILE * 3) + tid];

    // B tiles, already lane-swizzled: 2x global_load_b128 each
    const v16h b0 = *(const v16h*)(Bbuf + ((size_t)(t * 2 + 0) * 32 + lane) * 16);
    const v16h b1 = *(const v16h*)(Bbuf + ((size_t)(t * 2 + 1) * 32 + lane) * 16);
    if (t + 1 < N_KTILES)
      __builtin_prefetch(Bbuf + ((size_t)(t * 2 + 2) * 32 + lane) * 16, 0, 1);

    // generate A (16x32 f16): 16 weights per lane, ISA A-matrix layout.
    // Pure FMA chain seeded with lopac; radius mask = cndmask(mah, +inf)
    // on the exp2 input (exp2(-inf) == 0), neg folds into v_exp_f32.
    v16h a;
#pragma unroll
    for (int j = 0; j < 16; ++j) {
      const int kl = (j < 8) ? (kh8 + j) : (16 + kh8 + (j - 8));
      const float4 p0 = stage4[buf][kl * 3 + 0];   // m0 m1 m2 c00'
      const float4 p1 = stage4[buf][kl * 3 + 1];   // c01' c02' c11' c12'
      const float4 p2 = stage4[buf][kl * 3 + 2];   // c22' lopac - -
      const float d0 = vx - p0.x;
      const float d1 = vy - p0.y;
      const float d2 = vz - p0.z;
      const float q0 = d0 * d0, q1 = d1 * d1, q2 = d2 * d2;
      const float dist = q0 + q1 + q2;
      float mah = fmaf(p0.w, q0, p2.y);
      mah = fmaf(p1.z, q1, mah);
      mah = fmaf(p2.x, q2, mah);
      mah = fmaf(p1.x, d0 * d1, mah);
      mah = fmaf(p1.y, d0 * d2, mah);
      mah = fmaf(p1.w, d1 * d2, mah);
      mah = (dist < 16.0f) ? mah : __builtin_inff();
      a[j] = (_Float16)__builtin_exp2f(-mah);
    }

    acc0 = __builtin_amdgcn_wmma_f32_16x16x32_f16(
        false, a, false, b0, (short)0, acc0, false, false);
    acc1 = __builtin_amdgcn_wmma_f32_16x16x32_f16(
        false, a, false, b1, (short)0, acc1, false, false);

    __syncthreads();   // single barrier per tile (double-buffered stage)
  }

  // D layout: lane L, VGPR r -> (row M = r + 8*(L>>4), col N = L&15)
  const int c = lane & 15;
#pragma unroll
  for (int r = 0; r < 8; ++r) {
    const int v = vbase + kh8 + r;
    float val = acc0[r];
    if (c == 0) {
      // empty-gaussian term -> class 0 ; class 16 from acc1 column 0
      const float ex = vox[v * 3 + 0];
      const float ey = vox[v * 3 + 1];
      const float ez = vox[v * 3 + 2] - 2.2f;
      const float me = ex * ex * (1.0f / 1600.0f)
                     + ey * ey * (1.0f / 1600.0f)
                     + ez * ez * (1.0f / 10.24f);
      val += 0.5f * __expf(-0.5f * me);   // 5.0 * 0.1 * exp(...)
      out[(size_t)v * N_CLASSES + 16] = acc1[r];
    }
    out[(size_t)v * N_CLASSES + c] = val;
  }
}

// ---------------------------------------------------------------------------
extern "C" void kernel_launch(void* const* d_in, const int* in_sizes, int n_in,
                              void* d_out, int out_size, void* d_ws, size_t ws_size,
                              hipStream_t stream) {
  const float* props = (const float*)d_in[0];   // (1, 2048, 28) f32
  const float* vox   = (const float*)d_in[1];   // (80000, 3) f32
  float* out = (float*)d_out;                   // (1,100,100,8,17) f32

  float*    params = (float*)d_ws;                             // 98304 B
  _Float16* Bbuf   = (_Float16*)((char*)d_ws + PARAMS_BYTES);  // 131072 B

  prep_params_kernel<<<(N_GAUSS + 255) / 256, 256, 0, stream>>>(props, params);
  prep_bmat_kernel<<<(N_KTILES * 2 * 32 + 255) / 256, 256, 0, stream>>>(props, Bbuf);
  splat_kernel<<<N_VOX / 16 / 8, 256, 0, stream>>>(params, Bbuf, vox, out);
}
